// POPIAttnNet_90434831384607
// MI455X (gfx1250) — matrix operand order repaired and checked by reading, not compiled, not run on hardware
//
#include <hip/hip_runtime.h>
#include <hip/hip_bf16.h>

// ---------------------------------------------------------------------------
// POPI attention net on gfx1250: f16 WMMA GEMMs + async-LDS B staging +
// fused GRU gate math.
// ---------------------------------------------------------------------------

typedef __attribute__((ext_vector_type(16))) _Float16 v16h;
typedef __attribute__((ext_vector_type(8)))  float    v8f;
typedef __attribute__((ext_vector_type(4)))  int      v4i_t;

#define B_   4096
#define T_   9
#define F_   553
#define FP_  576        // F padded to multiple of 32 for WMMA K
#define R_   (B_ * T_)  // 36864 rows
#define U_   256
#define G3_  768        // 3*U gate width
#define D2_  512        // bidirectional feature width

// gfx1250 async global->LDS copy path (ASYNCcnt-tracked), with fallback.
#if defined(__has_builtin)
#if __has_builtin(__builtin_amdgcn_global_load_async_to_lds_b128) && \
    __has_builtin(__builtin_amdgcn_s_wait_asynccnt)
#define USE_ASYNC_LDS 1
#endif
#endif
#ifndef USE_ASYNC_LDS
#define USE_ASYNC_LDS 0
#endif

union AF { v16h v; uint4 q[2]; };
union CF { v8f  v; float f[8]; };

__device__ __forceinline__ float sigmoidf_(float x) { return 1.0f / (1.0f + __expf(-x)); }

// Copy 16 bytes global -> LDS. Async variant uses GLOBAL_LOAD_ASYNC_TO_LDS_B128.
__device__ __forceinline__ void stage16(const _Float16* __restrict__ src,
                                        _Float16* dst /* points into LDS */)
{
#if USE_ASYNC_LDS
  typedef __attribute__((address_space(1))) v4i_t* gq_t;
  typedef __attribute__((address_space(3))) v4i_t* lq_t;
  // Generic LDS address: low 32 bits are the LDS offset (ISA aperture rule).
  __builtin_amdgcn_global_load_async_to_lds_b128(
      (gq_t)(uintptr_t)src, (lq_t)(uint32_t)(uintptr_t)dst, 0, 0);
#else
  *(uint4*)dst = *(const uint4*)src;
#endif
}

// ---------------------------------------------------------------------------
// Generic WMMA GEMM: out[M,N] = A[M,Kpad](f16, row-major) @ Bfrag + bias
// Bfrag layout: (((ntile*KT + ktile)*32 + lane)*16 + e) halves, lane-contiguous.
// Each wave computes a 16(M) x 64(N) strip; block = 8 waves = 128 x 64 tile.
// B panel (64 cols x 32 K = 4KB) is double-buffered through LDS via async copy.
// MODE: 0 = f32 store, 1 = f16 store, 2 = f16 tanh, 3 = f16 relu
// ---------------------------------------------------------------------------
template <int MODE>
__global__ __launch_bounds__(256) void k_wmma_gemm(
    const _Float16* __restrict__ A, const _Float16* __restrict__ Bf,
    const float* __restrict__ bias, void* __restrict__ out, int Kpad, int N)
{
  __shared__ __align__(16) _Float16 ldsB[2][4 * 512];  // 2 x 4KB

  const int tid  = threadIdx.x;
  const int lane = tid & 31;
  const int wave = tid >> 5;
  const int m0   = blockIdx.x * 128 + wave * 16;
  const int n0   = blockIdx.y * 64;
  const int KT   = Kpad >> 5;
  const int kg   = lane >> 4;          // half-wave group
  const int row  = m0 + (lane & 15);

  // Cooperative B staging: thread copies 16B; ct = which of the 4 ntiles.
  const int ct = tid >> 6;
  const int co = (tid & 63) * 8;       // halves within the 512-half tile
  const _Float16* Bsrc = Bf + ((size_t)((n0 >> 4) + ct) * KT) * 512 + co;
  _Float16* Ldst[2] = { &ldsB[0][ct * 512 + co], &ldsB[1][ct * 512 + co] };

  CF c[4];
#pragma unroll
  for (int t = 0; t < 4; ++t)
#pragma unroll
    for (int v = 0; v < 8; ++v) c[t].f[v] = 0.0f;

  // A 16-bit 16x32 fragment: lane<16 -> K {kb..kb+15}, kb = kg*16
  const _Float16* Ap = A + (size_t)row * Kpad + kg * 16;

  stage16(Bsrc, Ldst[0]);  // prime pipeline with kt = 0

  for (int kt = 0; kt < KT; ++kt) {
    const int cur = kt & 1;
    if (kt + 1 < KT) stage16(Bsrc + (size_t)(kt + 1) * 512, Ldst[cur ^ 1]);
#if USE_ASYNC_LDS
    if (kt + 1 < KT) __builtin_amdgcn_s_wait_asynccnt(1);  // kt's panel done
    else             __builtin_amdgcn_s_wait_asynccnt(0);
#endif
    __syncthreads();  // panel kt visible to all 8 waves

    AF a;
    a.q[0] = *(const uint4*)(Ap);
    a.q[1] = *(const uint4*)(Ap + 8);
    Ap += 32;
    __builtin_prefetch(Ap, 0, 1);      // global_prefetch_b8 along the K stream

#pragma unroll
    for (int t = 0; t < 4; ++t) {
      const _Float16* bp = &ldsB[cur][t * 512 + lane * 16];
      AF b;
      b.q[0] = *(const uint4*)(bp);
      b.q[1] = *(const uint4*)(bp + 8);
      c[t].v = __builtin_amdgcn_wmma_f32_16x16x32_f16(
          /*neg_a=*/false, a.v, /*neg_b=*/false, b.v,
          /*c_mod=*/(short)0, c[t].v, /*reuse_a=*/false, /*reuse_b=*/false);
    }
    __syncthreads();  // all waves done reading buf[cur] before it is re-staged
  }

  // C/D layout: VGPR v -> M = v + 8*kg, N = lane&15
  const int colb = n0 + (lane & 15);
  const int rowb = m0 + kg * 8;
#pragma unroll
  for (int t = 0; t < 4; ++t) {
    const int col = colb + t * 16;
    const float bv = bias ? bias[col] : 0.0f;
#pragma unroll
    for (int v = 0; v < 8; ++v) {
      float val = c[t].f[v] + bv;
      const size_t idx = (size_t)(rowb + v) * N + col;
      if (MODE == 0) {
        ((float*)out)[idx] = val;
      } else {
        if (MODE == 2) val = tanhf(val);
        if (MODE == 3) val = fmaxf(val, 0.0f);
        ((_Float16*)out)[idx] = (_Float16)val;
      }
    }
  }
}

// ---------------------------------------------------------------------------
// Swizzle f32 weight [K,N] -> WMMA B-fragment f16 (zero-padded to KT*32 in K).
// B 16-bit 32x16 fragment: lane holds column N=lane&15, K = (lane>>4)*16 + e.
// ---------------------------------------------------------------------------
__global__ void k_swz(const float* __restrict__ W, _Float16* __restrict__ Fo,
                      int K, int N, int KT)
{
  const int i = blockIdx.x * 256 + threadIdx.x;
  const int total = (N >> 4) * KT * 512;
  if (i >= total) return;
  const int e    = i & 15;
  const int lane = (i >> 4) & 31;
  const int tile = i >> 9;
  const int kt   = tile % KT;
  const int nt   = tile / KT;
  const int n = (nt << 4) + (lane & 15);
  const int k = (kt << 5) + ((lane >> 4) << 4) + e;
  Fo[i] = (k < K) ? (_Float16)W[(size_t)k * N + n] : (_Float16)0.0f;
}

// Pack peptide f32 [R,553] -> f16 [R,576] (zero pad)
__global__ void k_packx(const float* __restrict__ X, _Float16* __restrict__ O)
{
  const int i = blockIdx.x * 256 + threadIdx.x;
  if (i >= R_ * FP_) return;
  const int r = i / FP_;
  const int k = i - r * FP_;
  O[i] = (k < F_) ? (_Float16)X[(size_t)r * F_ + k] : (_Float16)0.0f;
}

// Initialize GRU hidden state (src == nullptr -> zeros)
__global__ void k_inith(const float* __restrict__ src,
                        float* __restrict__ h32, _Float16* __restrict__ h16)
{
  const int i = blockIdx.x * 256 + threadIdx.x;
  const float v = src ? src[i] : 0.0f;
  h32[i] = v;
  h16[i] = (_Float16)v;
}

// Fused GRU gate math (Keras v2 reset_after): one thread per (b, j)
__global__ void k_gru_gate(const _Float16* __restrict__ xp,
                           const float* __restrict__ inner,
                           float* __restrict__ h32, _Float16* __restrict__ h16,
                           _Float16* __restrict__ Y, int timeIdx, int coloff)
{
  const int i = blockIdx.x * 256 + threadIdx.x;  // B_*U_ threads
  const int b = i >> 8;
  const int j = i & 255;
  const size_t xrow = ((size_t)b * T_ + timeIdx) * G3_;
  const float xz = (float)xp[xrow + j];
  const float xr = (float)xp[xrow + U_ + j];
  const float xh = (float)xp[xrow + 2 * U_ + j];
  const size_t irow = (size_t)b * G3_;
  const float iz = inner[irow + j];
  const float ir = inner[irow + U_ + j];
  const float ih = inner[irow + 2 * U_ + j];
  const float z  = sigmoidf_(xz + iz);
  const float r  = sigmoidf_(xr + ir);
  const float h  = h32[i];
  const float hh = tanhf(xh + r * ih);
  const float hn = z * h + (1.0f - z) * hh;
  h32[i] = hn;
  h16[i] = (_Float16)hn;
  Y[((size_t)b * T_ + timeIdx) * D2_ + coloff + j] = (_Float16)hn;
}

// Attention: scores = S@V + bV, softmax over T, context = sum w*Y3.
// One block per batch element.
__global__ __launch_bounds__(256) void k_attention(
    const _Float16* __restrict__ S, const _Float16* __restrict__ Y3,
    const float* __restrict__ V, const float* __restrict__ bV,
    float* __restrict__ out_attn, _Float16* __restrict__ ctx)
{
  __shared__ float red[256];
  __shared__ float wts[16];
  const int b = blockIdx.x;
  const int tid = threadIdx.x;

  for (int t = 0; t < T_; ++t) {
    const size_t base = ((size_t)b * T_ + t) * D2_;
    float p = (float)S[base + tid] * V[tid] +
              (float)S[base + 256 + tid] * V[256 + tid];
    red[tid] = p;
    __syncthreads();
    for (int s = 128; s > 0; s >>= 1) {
      if (tid < s) red[tid] += red[tid + s];
      __syncthreads();
    }
    if (tid == 0) wts[t] = red[0] + bV[0];
    __syncthreads();
  }
  if (tid == 0) {
    float mx = wts[0];
    for (int t = 1; t < T_; ++t) mx = fmaxf(mx, wts[t]);
    float e[T_], sum = 0.0f;
    for (int t = 0; t < T_; ++t) { e[t] = __expf(wts[t] - mx); sum += e[t]; }
    for (int t = 0; t < T_; ++t) {
      wts[t] = e[t] / sum;
      out_attn[(size_t)b * T_ + t] = wts[t];
    }
  }
  __syncthreads();
  for (int j = tid; j < D2_; j += 256) {
    float acc = 0.0f;
    for (int t = 0; t < T_; ++t)
      acc += wts[t] * (float)Y3[((size_t)b * T_ + t) * D2_ + j];
    ctx[(size_t)b * D2_ + j] = (_Float16)acc;
  }
}

// fc2: logits[b,o] = H1[b,:]@W[:,o] + b[o]   (N=2, scalar path)
__global__ void k_fc2(const _Float16* __restrict__ H1, const float* __restrict__ W,
                      const float* __restrict__ bias, float* __restrict__ logits)
{
  const int i = blockIdx.x * 256 + threadIdx.x;  // B_*2 threads
  const int b = i >> 1;
  const int o = i & 1;
  float acc = bias[o];
  for (int j = 0; j < D2_; ++j)
    acc += (float)H1[(size_t)b * D2_ + j] * W[j * 2 + o];
  logits[i] = acc;
}

// ---------------------------------------------------------------------------

extern "C" void kernel_launch(void* const* d_in, const int* in_sizes, int n_in,
                              void* d_out, int out_size, void* d_ws, size_t ws_size,
                              hipStream_t stream)
{
  (void)in_sizes; (void)n_in; (void)out_size; (void)ws_size;

  const float* peptide = (const float*)d_in[0];
  const float* hidden  = (const float*)d_in[1];
  const float *gW[6], *gU[6], *gb[6];
  for (int i = 0; i < 6; ++i) {
    gW[i] = (const float*)d_in[2 + 3 * i];
    gU[i] = (const float*)d_in[3 + 3 * i];
    gb[i] = (const float*)d_in[4 + 3 * i];
  }
  const float* attW1 = (const float*)d_in[20];
  const float* attb1 = (const float*)d_in[21];
  const float* attV  = (const float*)d_in[22];
  const float* attbV = (const float*)d_in[23];
  const float* fc1W  = (const float*)d_in[24];
  const float* fc1b  = (const float*)d_in[25];
  const float* fc2W  = (const float*)d_in[26];
  const float* fc2b  = (const float*)d_in[27];

  char* ws = (char*)d_ws;
  size_t off = 0;
  auto alloc = [&](size_t bytes) -> char* {
    char* p = ws + off;
    off = (off + bytes + 255) & ~(size_t)255;
    return p;
  };

  _Float16* Xin = (_Float16*)alloc((size_t)R_ * FP_ * 2);
  _Float16* Y[3];
  for (int l = 0; l < 3; ++l) Y[l] = (_Float16*)alloc((size_t)R_ * D2_ * 2);
  _Float16* S = (_Float16*)alloc((size_t)R_ * D2_ * 2);
  _Float16* XP[2];
  for (int d = 0; d < 2; ++d) XP[d] = (_Float16*)alloc((size_t)R_ * G3_ * 2);
  float* INN[2];
  for (int d = 0; d < 2; ++d) INN[d] = (float*)alloc((size_t)B_ * G3_ * 4);
  float* H32[2];
  for (int d = 0; d < 2; ++d) H32[d] = (float*)alloc((size_t)B_ * U_ * 4);
  _Float16* H16[2];
  for (int d = 0; d < 2; ++d) H16[d] = (_Float16*)alloc((size_t)B_ * U_ * 2);
  _Float16* CTX = (_Float16*)alloc((size_t)B_ * D2_ * 2);
  _Float16* H1  = (_Float16*)alloc((size_t)B_ * D2_ * 2);

  const int KTw[6] = {FP_ / 32, FP_ / 32, 16, 16, 16, 16};
  const int Kin[6] = {F_, F_, D2_, D2_, D2_, D2_};
  _Float16 *Wf[6], *Uf[6];
  for (int i = 0; i < 6; ++i) {
    Wf[i] = (_Float16*)alloc((size_t)KTw[i] * (G3_ / 16) * 512 * 2);
    Uf[i] = (_Float16*)alloc((size_t)(U_ / 32) * (G3_ / 16) * 512 * 2);
  }
  _Float16* A1f = (_Float16*)alloc((size_t)16 * 32 * 512 * 2);
  _Float16* F1f = (_Float16*)alloc((size_t)16 * 32 * 512 * 2);

  auto swz = [&](const float* W, _Float16* Fo, int K, int N, int KT) {
    const int total = (N >> 4) * KT * 512;
    k_swz<<<(total + 255) / 256, 256, 0, stream>>>(W, Fo, K, N, KT);
  };

  for (int i = 0; i < 6; ++i) {
    swz(gW[i], Wf[i], Kin[i], G3_, KTw[i]);
    swz(gU[i], Uf[i], U_, G3_, U_ / 32);
  }
  swz(attW1, A1f, D2_, D2_, 16);
  swz(fc1W, F1f, D2_, D2_, 16);

  k_packx<<<(R_ * FP_ + 255) / 256, 256, 0, stream>>>(peptide, Xin);

  // --- 3 stacked bi-GRU layers ---
  for (int l = 0; l < 3; ++l) {
    const _Float16* Xl = (l == 0) ? Xin : Y[l - 1];
    const int Kpad = (l == 0) ? FP_ : D2_;
    for (int d = 0; d < 2; ++d) {
      const int ci = l * 2 + d;
      dim3 g(R_ / 128, G3_ / 64);
      // xp = X @ W + b_in   (f16 out)
      k_wmma_gemm<1><<<g, 256, 0, stream>>>(Xl, Wf[ci], gb[ci], XP[d], Kpad, G3_);
      k_inith<<<(B_ * U_) / 256, 256, 0, stream>>>(l == 0 ? hidden : nullptr,
                                                   H32[d], H16[d]);
    }
    for (int s = 0; s < T_; ++s) {
      for (int d = 0; d < 2; ++d) {
        const int ci = l * 2 + d;
        dim3 g(B_ / 128, G3_ / 64);
        // inner = h @ U + b_rec   (f32 out)
        k_wmma_gemm<0><<<g, 256, 0, stream>>>(H16[d], Uf[ci], gb[ci] + G3_,
                                              INN[d], U_, G3_);
        const int timeIdx = (d == 0) ? s : (T_ - 1 - s);
        k_gru_gate<<<(B_ * U_) / 256, 256, 0, stream>>>(
            XP[d], INN[d], H32[d], H16[d], Y[l], timeIdx, d * U_);
      }
    }
  }

  // --- attention + head ---
  {
    dim3 g(R_ / 128, D2_ / 64);
    k_wmma_gemm<2><<<g, 256, 0, stream>>>(Y[2], A1f, attb1, S, D2_, D2_);
  }
  float* logits = (float*)d_out;
  float* attn   = logits + (size_t)B_ * 2;
  k_attention<<<B_, 256, 0, stream>>>(S, Y[2], attV, attbV, attn, CTX);
  {
    dim3 g(B_ / 128, D2_ / 64);
    k_wmma_gemm<3><<<g, 256, 0, stream>>>(CTX, F1f, fc1b, H1, D2_, D2_);
  }
  k_fc2<<<(B_ * 2) / 256, 256, 0, stream>>>(H1, fc2W, fc2b, logits);
}
